// DGLJTNNEncoder_69002944577982
// MI455X (gfx1250) — matrix operand order repaired and checked by reading, not compiled, not run on hardware
//
#include <hip/hip_runtime.h>
#include <hip/hip_bf16.h>

// Fused bottom-up TreeGRU for the DGL-JTNN encoder on gfx1250.
// Down-pass groups are provably dead w.r.t. the returned root vectors, so only
// the 10 bottom-up levels + root readout are computed.
//
// Workspace layout (needs ~100.9 MB):
//   x16  : f16 [N][128]  node embeddings
//   m16  : f16 [N][128]  per-node upward message
//   rm16 : f16 [N][128]  per-node r*m
//   4 packed f16 weight matrices (256x128 each) in WMMA B-fragment layout

typedef __attribute__((ext_vector_type(16))) _Float16 v16h;
typedef __attribute__((ext_vector_type(8)))  float    v8f;

#define TREES  64
#define NPT    2047
#define HDIM   128
#define NNODES (TREES * NPT)      // 131008
#define TDEPTH 10

__device__ __forceinline__ float sigmoidf_(float x) {
  return 1.0f / (1.0f + __expf(-x));
}

// A-fragment (16x32 f16) from row-major LDS staging [16][256], K-tile kt.
// ISA layout: lane<16 -> M=lane, halves 0..7 => K=kbase+0..7, 8..15 => K=kbase+16..23
//             lane>=16 -> kbase=8.
__device__ __forceinline__ v16h load_a_frag(const _Float16* stg, int lane, int kt) {
  const int row   = lane & 15;
  const int kbase = (lane & 16) ? 8 : 0;
  const _Float16* p = stg + row * 256 + kt * 32 + kbase;
  v16h a;
#pragma unroll
  for (int j = 0; j < 8; ++j) { a[j] = p[j]; a[j + 8] = p[16 + j]; }
  return a;
}

// B fragments are pre-packed contiguously: frag (kt,nt), 32 lanes x 16 halves.
__device__ __forceinline__ v16h load_b_frag(const _Float16* w, int lane, int kt, int nt) {
  return ((const v16h*)w)[(size_t)((kt * 8 + nt) * 32 + lane)];
}

// ---------------------------------------------------------------------------
// x16[node][c] = (f16) emb[wid[node]][c]
__global__ void __launch_bounds__(256)
gather_x_kernel(const int* __restrict__ wid, const float* __restrict__ emb,
                _Float16* __restrict__ x16) {
  int idx = blockIdx.x * 256 + threadIdx.x;
  if (idx >= NNODES * HDIM) return;
  int node = idx >> 7, col = idx & 127;
  x16[idx] = (_Float16)emb[wid[node] * HDIM + col];
}

// ---------------------------------------------------------------------------
// Pack a 256x128 f32 weight (rows 0..127 from src0, 128..255 from src1) into
// WMMA B-fragment f16 layout: K = kt*32 + (lane&16 ? 8:0) + (j<8 ? j : j+8), N = nt*16 + (lane&15).
__global__ void __launch_bounds__(256)
pack_weight_kernel(const float* __restrict__ src0, const float* __restrict__ src1,
                   _Float16* __restrict__ dst) {
  int idx  = blockIdx.x * 256 + threadIdx.x;   // 0 .. 32767
  int j    = idx & 15;
  int lane = (idx >> 4) & 31;
  int f    = idx >> 9;                          // 0..63
  int nt   = f & 7, kt = f >> 3;
  int K = kt * 32 + ((lane & 16) ? 8 : 0) + ((j < 8) ? j : j + 8);
  int N = nt * 16 + (lane & 15);
  const float* s = (K < 128) ? (src0 + K * HDIM + N) : (src1 + (K - 128) * HDIM + N);
  dst[idx] = (_Float16)(*s);
}

// ---------------------------------------------------------------------------
// One level of the bottom-up TreeGRU. One wave (32 threads) per 16-node tile.
// Three fused K=256 GEMMs (z, h~, r) via v_wmma_f32_16x16x32_f16.
// LEAF is a template parameter so every k-loop has a compile-time trip count
// and the A-fragment array is never dynamically indexed (no scratch spills).
template <bool LEAF>
__global__ void __launch_bounds__(32)
tree_gru_level(const _Float16* __restrict__ x16,
               _Float16* __restrict__ m16,
               _Float16* __restrict__ rm16,
               const _Float16* __restrict__ Wz,
               const _Float16* __restrict__ Wh,
               const _Float16* __restrict__ WrUr,
               const float* __restrict__ bz,
               const float* __restrict__ bh,
               const float* __restrict__ br,
               int d) {
  __shared__ _Float16 stg[16 * 256];   // row-major A staging [x | other]
  __shared__ float sS[16 * HDIM];      // s (children message sum)
  __shared__ float sZ[16 * HDIM];      // z gate
  __shared__ float sM[16 * HDIM];      // new message m
  __shared__ int s_node[16], s_parent[16], s_c0[16];

  constexpr int KTZH = LEAF ? 4 : 8;   // k-tiles for the z / h GEMMs

  const int lane = threadIdx.x;

  if (lane < 16) {
    int r     = blockIdx.x * 16 + lane;
    int b     = r >> d;
    int off   = r & ((1 << d) - 1);
    int local = (1 << d) - 1 + off;
    s_node[lane]   = b * NPT + local;
    s_parent[lane] = b * NPT + ((local - 1) >> 1);
    s_c0[lane]     = b * NPT + 2 * local + 1;
  }
  __syncthreads();

  // Phase 1: stage A = [x | s]; keep s in f32 for the gate blend.
  for (int t = lane; t < 16 * HDIM; t += 32) {
    int i = t >> 7, col = t & 127;
    float xv = (float)x16[(size_t)s_node[i] * HDIM + col];
    float sv = 0.0f;
    if (!LEAF) {
      int c0 = s_c0[i];
      sv = (float)m16[(size_t)c0 * HDIM + col] + (float)m16[(size_t)(c0 + 1) * HDIM + col];
    }
    stg[i * 256 + col]       = (_Float16)xv;
    if (!LEAF) stg[i * 256 + 128 + col] = (_Float16)sv;
    sS[t] = sv;
  }
  __syncthreads();

  const int rowoff = (lane & 16) ? 8 : 0;
  const int coll   = lane & 15;

  v16h a[8];
#pragma unroll
  for (int kt = 0; kt < KTZH; ++kt) a[kt] = load_a_frag(stg, lane, kt);

  // GEMM 1: z = sigmoid([x|s] @ Wz + bz)
  for (int nt = 0; nt < 8; ++nt) {
    int col  = nt * 16 + coll;
    float bv = bz[col];
    v8f acc;
#pragma unroll
    for (int e = 0; e < 8; ++e) acc[e] = bv;
#pragma unroll
    for (int kt = 0; kt < KTZH; ++kt) {
      v16h bf = load_b_frag(Wz, lane, kt, nt);
      acc = __builtin_amdgcn_wmma_f32_16x16x32_f16(false, a[kt], false, bf,
                                                   (short)0, acc, false, false);
    }
#pragma unroll
    for (int e = 0; e < 8; ++e) sZ[(e + rowoff) * HDIM + col] = sigmoidf_(acc[e]);
  }
  __syncthreads();

  // Phase 2: restage bottom half with arm = sum of children rm.
  if (!LEAF) {
    for (int t = lane; t < 16 * HDIM; t += 32) {
      int i = t >> 7, col = t & 127;
      int c0 = s_c0[i];
      float av = (float)rm16[(size_t)c0 * HDIM + col] +
                 (float)rm16[(size_t)(c0 + 1) * HDIM + col];
      stg[i * 256 + 128 + col] = (_Float16)av;
    }
    __syncthreads();
#pragma unroll
    for (int kt = 4; kt < 8; ++kt) a[kt] = load_a_frag(stg, lane, kt);
  }

  // GEMM 2: h~ = tanh([x|arm] @ Wh + bh); m = (1-z)s + z*h~
  for (int nt = 0; nt < 8; ++nt) {
    int col  = nt * 16 + coll;
    float bv = bh[col];
    v8f acc;
#pragma unroll
    for (int e = 0; e < 8; ++e) acc[e] = bv;
#pragma unroll
    for (int kt = 0; kt < KTZH; ++kt) {
      v16h bf = load_b_frag(Wh, lane, kt, nt);
      acc = __builtin_amdgcn_wmma_f32_16x16x32_f16(false, a[kt], false, bf,
                                                   (short)0, acc, false, false);
    }
#pragma unroll
    for (int e = 0; e < 8; ++e) {
      int row  = e + rowoff;
      float ht = tanhf(acc[e]);
      float z  = sZ[row * HDIM + col];
      float s  = sS[row * HDIM + col];
      float mv = (1.0f - z) * s + z * ht;
      sM[row * HDIM + col]       = mv;
      stg[row * 256 + 128 + col] = (_Float16)mv;                   // A bottom := m
      m16[(size_t)s_node[row] * HDIM + col] = (_Float16)mv;
    }
  }
  __syncthreads();

  // Phase 3: top half := x[parent]
  for (int t = lane; t < 16 * HDIM; t += 32) {
    int i = t >> 7, col = t & 127;
    stg[i * 256 + col] = x16[(size_t)s_parent[i] * HDIM + col];
  }
  __syncthreads();
#pragma unroll
  for (int kt = 0; kt < 8; ++kt) a[kt] = load_a_frag(stg, lane, kt);

  // GEMM 3: r = sigmoid([x_parent|m] @ [Wr;Ur] + br); rm = r*m
  for (int nt = 0; nt < 8; ++nt) {
    int col  = nt * 16 + coll;
    float bv = br[col];
    v8f acc;
#pragma unroll
    for (int e = 0; e < 8; ++e) acc[e] = bv;
#pragma unroll
    for (int kt = 0; kt < 8; ++kt) {
      v16h bf = load_b_frag(WrUr, lane, kt, nt);
      acc = __builtin_amdgcn_wmma_f32_16x16x32_f16(false, a[kt], false, bf,
                                                   (short)0, acc, false, false);
    }
#pragma unroll
    for (int e = 0; e < 8; ++e) {
      int row  = e + rowoff;
      float rv = sigmoidf_(acc[e]);
      float rm = rv * sM[row * HDIM + col];
      rm16[(size_t)s_node[row] * HDIM + col] = (_Float16)rm;
    }
  }
}

// ---------------------------------------------------------------------------
// Root readout: out[b] = relu([x_root | m_c1 + m_c2] @ Wg + bg)
__global__ void __launch_bounds__(32)
tree_gru_root(const _Float16* __restrict__ x16, const _Float16* __restrict__ m16,
              const _Float16* __restrict__ Wg, const float* __restrict__ bg,
              float* __restrict__ out) {
  __shared__ _Float16 stg[16 * 256];
  const int lane = threadIdx.x;

  for (int t = lane; t < 16 * HDIM; t += 32) {
    int i = t >> 7, col = t & 127;
    int node = (blockIdx.x * 16 + i) * NPT;
    stg[i * 256 + col] = x16[(size_t)node * HDIM + col];
    float ms = (float)m16[(size_t)(node + 1) * HDIM + col] +
               (float)m16[(size_t)(node + 2) * HDIM + col];
    stg[i * 256 + 128 + col] = (_Float16)ms;
  }
  __syncthreads();

  v16h a[8];
#pragma unroll
  for (int kt = 0; kt < 8; ++kt) a[kt] = load_a_frag(stg, lane, kt);

  const int rowoff = (lane & 16) ? 8 : 0;
  const int coll   = lane & 15;
  for (int nt = 0; nt < 8; ++nt) {
    int col  = nt * 16 + coll;
    float bv = bg[col];
    v8f acc;
#pragma unroll
    for (int e = 0; e < 8; ++e) acc[e] = bv;
#pragma unroll
    for (int kt = 0; kt < 8; ++kt) {
      v16h bf = load_b_frag(Wg, lane, kt, nt);
      acc = __builtin_amdgcn_wmma_f32_16x16x32_f16(false, a[kt], false, bf,
                                                   (short)0, acc, false, false);
    }
#pragma unroll
    for (int e = 0; e < 8; ++e) {
      int b = blockIdx.x * 16 + e + rowoff;
      out[(size_t)b * HDIM + col] = fmaxf(acc[e], 0.0f);
    }
  }
}

// ---------------------------------------------------------------------------
extern "C" void kernel_launch(void* const* d_in, const int* in_sizes, int n_in,
                              void* d_out, int out_size, void* d_ws, size_t ws_size,
                              hipStream_t stream) {
  const int*   wid  = (const int*)d_in[0];
  const float* emb  = (const float*)d_in[1];
  const float* Wz_w = (const float*)d_in[2];
  const float* Wz_b = (const float*)d_in[3];
  const float* Wr_w = (const float*)d_in[4];
  const float* Ur_w = (const float*)d_in[5];
  const float* Ur_b = (const float*)d_in[6];
  const float* Wh_w = (const float*)d_in[7];
  const float* Wh_b = (const float*)d_in[8];
  const float* Wg_w = (const float*)d_in[9];
  const float* Wg_b = (const float*)d_in[10];
  float* out = (float*)d_out;

  char* ws = (char*)d_ws;
  const size_t xbytes = (size_t)NNODES * HDIM * sizeof(_Float16);  // 33.5 MB
  _Float16* x16  = (_Float16*)(ws);
  _Float16* m16  = (_Float16*)(ws + xbytes);
  _Float16* rm16 = (_Float16*)(ws + 2 * xbytes);
  _Float16* Wzp  = (_Float16*)(ws + 3 * xbytes);
  _Float16* Whp  = Wzp + 256 * HDIM;
  _Float16* Wrp  = Whp + 256 * HDIM;
  _Float16* Wgp  = Wrp + 256 * HDIM;

  gather_x_kernel<<<(NNODES * HDIM + 255) / 256, 256, 0, stream>>>(wid, emb, x16);
  pack_weight_kernel<<<128, 256, 0, stream>>>(Wz_w, Wz_w + 128 * HDIM, Wzp);
  pack_weight_kernel<<<128, 256, 0, stream>>>(Wh_w, Wh_w + 128 * HDIM, Whp);
  pack_weight_kernel<<<128, 256, 0, stream>>>(Wr_w, Ur_w,              Wrp);
  pack_weight_kernel<<<128, 256, 0, stream>>>(Wg_w, Wg_w + 128 * HDIM, Wgp);

  // Leaf level (d = 10): children are absent, K=128 GEMMs.
  {
    int rows = TREES << TDEPTH;
    tree_gru_level<true><<<rows / 16, 32, 0, stream>>>(x16, m16, rm16, Wzp, Whp, Wrp,
                                                       Wz_b, Wh_b, Ur_b, TDEPTH);
  }
  // Interior levels (d = 9..1): full K=256 GEMMs.
  for (int d = TDEPTH - 1; d >= 1; --d) {
    int rows = TREES << d;
    tree_gru_level<false><<<rows / 16, 32, 0, stream>>>(x16, m16, rm16, Wzp, Whp, Wrp,
                                                        Wz_b, Wh_b, Ur_b, d);
  }
  tree_gru_root<<<TREES / 16, 32, 0, stream>>>(x16, m16, Wgp, Wg_b, out);
}
